// PixelVectorQuantizer_67989332296383
// MI455X (gfx1250) — matrix operand order repaired and checked by reading, not compiled, not run on hardware
//
#include <hip/hip_runtime.h>
#include <hip/hip_bf16.h>

typedef __attribute__((ext_vector_type(16))) __bf16 v16bf;
typedef __attribute__((ext_vector_type(8)))  __bf16 v8bf;
typedef __attribute__((ext_vector_type(8)))  float  v8f;

#define B_   16
#define C_   256
#define H_   64
#define W_   64
#define K_   1024
#define HW_  (H_ * W_)          // 4096
#define NPIX (B_ * HW_)         // 65536
#define SROW 72                 // LDS row stride (bf16 elems) for 64-channel phase

static __device__ __forceinline__ unsigned short f32_to_bf16_rne(float f) {
    union { float f; unsigned int u; } v; v.f = f;
    unsigned int u = v.u;
    unsigned int r = u + 0x7FFFu + ((u >> 16) & 1u);
    if ((u & 0x7F800000u) == 0x7F800000u && (u & 0x007FFFFFu)) r = u | 0x00400000u; // NaN
    return (unsigned short)(r >> 16);
}

// ---- prep: codebook fp32 -> bf16 ----
__global__ __launch_bounds__(256) void vq_prep_cb(const float* __restrict__ cb,
                                                  unsigned short* __restrict__ cb_bf) {
    int i = blockIdx.x * 256 + threadIdx.x;   // over K_*C_
    cb_bf[i] = f32_to_bf16_rne(cb[i]);
}

// ---- prep: per-code squared norms (fp32) ----
__global__ __launch_bounds__(256) void vq_prep_norm(const float* __restrict__ cb,
                                                    float* __restrict__ norms) {
    __shared__ float red[256];
    int k = blockIdx.x, c = threadIdx.x;
    float v = cb[(size_t)k * C_ + c];
    red[c] = v * v;
    __syncthreads();
    for (int s = 128; s > 0; s >>= 1) {
        if (c < s) red[c] += red[c + s];
        __syncthreads();
    }
    if (c == 0) norms[k] = red[0];
}

// ---- fused bf16-WMMA GEMM + argmin over 1024 codes ----
// block = 256 threads (8 waves) handles 256 pixels; each wave owns TWO 16-pixel
// tiles -> every B fragment feeds two independent WMMA accumulator chains.
__global__ __launch_bounds__(256) void vq_assign(const float* __restrict__ z,
                                                 const unsigned short* __restrict__ cb_bf,
                                                 const float* __restrict__ norms,
                                                 int* __restrict__ out_idx) {
    __shared__ __align__(16) unsigned short sA[256 * SROW];   // 36864 B
    __shared__ __align__(16) float sNorm[K_];                 //  4096 B

    const int tid  = threadIdx.x;
    const int wave = tid >> 5;
    const int lane = tid & 31;
    const int col  = lane & 15;   // N (code) column within tile
    const int half = lane >> 4;   // lane-group selector per ISA layouts

    const int px0 = blockIdx.x * 256;
    const int b   = px0 >> 12;            // 4096 pixels per image, 256 | 4096
    const int hw0 = px0 & 4095;
    const float* zb = z + (size_t)b * C_ * HW_ + hw0;

    // stage code norms to LDS once (covered by first phase barrier)
    {
        const float4 nv = *(const float4*)(norms + tid * 4);
        *(float4*)(&sNorm[tid * 4]) = nv;
    }

    // Persistent A fragments: 2 tiles x 16 pixels x 256 channels (bf16).
    v16bf a0[8], a1[8];
    const int row0 = wave * 16 + col;         // tile 0 pixel row
    const int row1 = 128 + wave * 16 + col;   // tile 1 pixel row

    #pragma unroll
    for (int ph = 0; ph < 4; ++ph) {
        const int cbase = ph * 64;
        // stage 64 channels x 256 pixels: float4 global loads (coalesced),
        // fp32 -> bf16, transpose via b16 DS stores.
        #pragma unroll 4
        for (int it = 0; it < 16; ++it) {
            int slot = it * 256 + tid;          // 0..4095
            int c    = slot >> 6;               // 0..63
            int px   = (slot & 63) << 2;        // 0,4,...,252
            const float4 v = *(const float4*)(zb + (size_t)(cbase + c) * HW_ + px);
            sA[(px + 0) * SROW + c] = f32_to_bf16_rne(v.x);
            sA[(px + 1) * SROW + c] = f32_to_bf16_rne(v.y);
            sA[(px + 2) * SROW + c] = f32_to_bf16_rne(v.z);
            sA[(px + 3) * SROW + c] = f32_to_bf16_rne(v.w);
        }
        __syncthreads();
        // A fragments per ISA 16-bit 16x32 layout:
        // lanes<16: K {kb..kb+7, kb+16..kb+23}; lanes>=16: K {kb+8..+15, kb+24..+31}
        #pragma unroll
        for (int sl = 0; sl < 2; ++sl) {
            const int off = sl * 32 + half * 8;
            v8bf l0 = *(const v8bf*)&sA[row0 * SROW + off];
            v8bf h0 = *(const v8bf*)&sA[row0 * SROW + off + 16];
            v8bf l1 = *(const v8bf*)&sA[row1 * SROW + off];
            v8bf h1 = *(const v8bf*)&sA[row1 * SROW + off + 16];
            a0[ph * 2 + sl] = __builtin_shufflevector(
                l0, h0, 0, 1, 2, 3, 4, 5, 6, 7, 8, 9, 10, 11, 12, 13, 14, 15);
            a1[ph * 2 + sl] = __builtin_shufflevector(
                l1, h1, 0, 1, 2, 3, 4, 5, 6, 7, 8, 9, 10, 11, 12, 13, 14, 15);
        }
        __syncthreads();   // allow restaging of next phase
    }

    float best0[8], best1[8];
    int   bidx0[8], bidx1[8];
    #pragma unroll
    for (int j = 0; j < 8; ++j) {
        best0[j] = 3.4e38f; bidx0[j] = 0;
        best1[j] = 3.4e38f; bidx1[j] = 0;
    }

    // 64 code tiles of 16 codes; full-C accumulation = 8 WMMAs per tile per chain.
    for (int ct = 0; ct < 64; ++ct) {
        const int code = ct * 16 + col;
        // B matches A: lanes<16 hold K {0..7, 16..23}, lanes>=16 K {8..15, 24..31}.
        const unsigned short* brow = cb_bf + (size_t)code * C_ + half * 8;
        v8f acc0 = {}, acc1 = {};
        #pragma unroll
        for (int s = 0; s < 8; ++s) {
            v8bf lo = *(const v8bf*)(brow + s * 32);
            v8bf hi = *(const v8bf*)(brow + s * 32 + 16);
            v16bf bf = __builtin_shufflevector(
                lo, hi, 0, 1, 2, 3, 4, 5, 6, 7, 8, 9, 10, 11, 12, 13, 14, 15);
            acc0 = __builtin_amdgcn_wmma_f32_16x16x32_bf16(
                false, a0[s], false, bf, (short)0, acc0, false, false);
            acc1 = __builtin_amdgcn_wmma_f32_16x16x32_bf16(
                false, a1[s], false, bf, (short)0, acc1, false, false);
        }
        // dist = ||cb||^2 - 2*z.cb  (||z||^2 constant per pixel -> dropped)
        const float nrm = sNorm[code];
        #pragma unroll
        for (int j = 0; j < 8; ++j) {
            float d0 = nrm - 2.0f * acc0[j];
            float d1 = nrm - 2.0f * acc1[j];
            if (d0 < best0[j]) { best0[j] = d0; bidx0[j] = code; }
            if (d1 < best1[j]) { best1[j] = d1; bidx1[j] = code; }
        }
    }

    // Cross-lane argmin over 16 code columns (C/D layout: lane group of 16 shares
    // pixel rows M = j + 8*half). Tie-break on smaller index (argmin semantics).
    #pragma unroll
    for (int j = 0; j < 8; ++j) {
        float v0 = best0[j]; int i0 = bidx0[j];
        float v1 = best1[j]; int i1 = bidx1[j];
        #pragma unroll
        for (int off = 1; off < 16; off <<= 1) {
            float ov0 = __shfl_xor(v0, off, 32); int oi0 = __shfl_xor(i0, off, 32);
            float ov1 = __shfl_xor(v1, off, 32); int oi1 = __shfl_xor(i1, off, 32);
            if (ov0 < v0 || (ov0 == v0 && oi0 < i0)) { v0 = ov0; i0 = oi0; }
            if (ov1 < v1 || (ov1 == v1 && oi1 < i1)) { v1 = ov1; i1 = oi1; }
        }
        if (col == 0) {
            int pix = px0 + wave * 16 + half * 8 + j;
            out_idx[pix]       = i0;
            out_idx[pix + 128] = i1;
        }
    }
}

// ---- gather: zq[b][c][hw] = codebook[idx][c], plus float indices ----
// one thread per 4 consecutive pixels -> b128 coalesced zq stores.
__global__ __launch_bounds__(256) void vq_gather(const int* __restrict__ idxbuf,
                                                 const float* __restrict__ cb,
                                                 float* __restrict__ zq,
                                                 float* __restrict__ oidx) {
    const int p4 = (blockIdx.x * 256 + threadIdx.x) * 4;   // 4 pixels per thread
    const int4 idx = *(const int4*)(idxbuf + p4);
    const int b  = p4 >> 12;
    const int hw = p4 & 4095;
    float* dst = zq + (size_t)b * C_ * HW_ + hw;
    const float* s0 = cb + (size_t)idx.x * C_;   // codebook is L2-resident (1 MB)
    const float* s1 = cb + (size_t)idx.y * C_;
    const float* s2 = cb + (size_t)idx.z * C_;
    const float* s3 = cb + (size_t)idx.w * C_;
    #pragma unroll 4
    for (int c = 0; c < C_; ++c) {
        float4 o; o.x = s0[c]; o.y = s1[c]; o.z = s2[c]; o.w = s3[c];
        *(float4*)(dst + (size_t)c * HW_) = o;   // 16B stores, coalesced across lanes
    }
    float4 fo; fo.x = (float)idx.x; fo.y = (float)idx.y;
    fo.z = (float)idx.z; fo.w = (float)idx.w;
    *(float4*)(oidx + p4) = fo;
}

extern "C" void kernel_launch(void* const* d_in, const int* in_sizes, int n_in,
                              void* d_out, int out_size, void* d_ws, size_t ws_size,
                              hipStream_t stream) {
    const float* z  = (const float*)d_in[0];   // [B,C,H,W] fp32
    const float* cb = (const float*)d_in[1];   // [K,C]     fp32

    // workspace layout: bf16 codebook (512KB) | norms (4KB) | int indices (256KB)
    unsigned short* cb_bf = (unsigned short*)d_ws;
    float* norms = (float*)((char*)d_ws + (size_t)K_ * C_ * 2);
    int*   idxbf = (int*)((char*)d_ws + (size_t)K_ * C_ * 2 + (size_t)K_ * 4);

    float* zq   = (float*)d_out;                           // [B,C,H,W]
    float* oidx = (float*)d_out + (size_t)B_ * C_ * HW_;   // [B,1,H,W]

    vq_prep_cb  <<<(K_ * C_) / 256, 256, 0, stream>>>(cb, cb_bf);
    vq_prep_norm<<<K_,              256, 0, stream>>>(cb, norms);
    vq_assign   <<<NPIX / 256,      256, 0, stream>>>(z, cb_bf, norms, idxbf);
    vq_gather   <<<NPIX / (4*256),  256, 0, stream>>>(idxbf, cb, zq, oidx);
}
